// Attention_82420422410365
// MI455X (gfx1250) — matrix operand order repaired and checked by reading, not compiled, not run on hardware
//
#include <hip/hip_runtime.h>
#include <hip/hip_bf16.h>
#include <math.h>

// ---------------------------------------------------------------------------
// Bahdanau attention for MI455X (gfx1250, wave32, WMMA).
//   energies[b,t] = sum_k tanh( (hidden@W1^T)[b,k] + (enc@W2^T)[b,t,k] + bias[k] ) * v[k]
//   out[b,0,t]    = softmax_t(energies[b,t])
// Dominant GEMM (M=65536, N=1024, K=1024) runs as v_wmma_f32_16x16x32_bf16
// with f32 accumulation. Per wave: 2 M-tiles (B fragment reuse) x 2 split-K
// accumulators (4 independent WMMA chains). 8 waves/WG via N-range split,
// partial energies summed in the softmax pass.
// ---------------------------------------------------------------------------

typedef __attribute__((ext_vector_type(16))) __bf16 v16bf;
typedef __attribute__((ext_vector_type(8)))  float  v8f;

#define B_DIM 32
#define T_DIM 2048
#define H_DIM 1024   // reduction dim of the big GEMM
#define K_DIM 1024   // output dim (pre-tanh features)
#define TQ    128    // t-rows per workgroup (8 m-tiles of 16 rows)

__device__ __forceinline__ unsigned short f2bf(float f) {
    unsigned u = __float_as_uint(f);
    unsigned r = u + 0x7FFFu + ((u >> 16) & 1u);   // round-to-nearest-even
    return (unsigned short)(r >> 16);
}
__device__ __forceinline__ unsigned pack2(float lo, float hi) {
    return (unsigned)f2bf(lo) | ((unsigned)f2bf(hi) << 16);
}

// branch-free tanh: v_exp_f32 + v_rcp, no EXEC divergence
__device__ __forceinline__ float fast_tanh(float x) {
    float ax = fabsf(x);
    float e  = __expf(-2.0f * ax);
    float t  = (1.0f - e) * __frcp_rn(1.0f + e);
    return copysignf(t, x);
}

// ---- prep 1: W2 (= W[:, H:]) -> bf16, k-major [k][h] -----------------------
__global__ void w2_to_bf16_kernel(const float* __restrict__ W,
                                  unsigned short* __restrict__ W2b) {
    int idx = blockIdx.x * 256 + threadIdx.x;      // K_DIM*H_DIM threads
    int k = idx >> 10, h = idx & 1023;
    W2b[idx] = f2bf(W[(size_t)k * (2 * H_DIM) + H_DIM + h]);
}

// ---- prep 2: s1[b,k] = hidden[b,:] . W[k, :H] + bias[k] --------------------
__global__ void s1_kernel(const float* __restrict__ hidden,
                          const float* __restrict__ W,
                          const float* __restrict__ bias,
                          float* __restrict__ s1) {
    int idx = blockIdx.x * 256 + threadIdx.x;      // B_DIM*K_DIM threads
    int b = idx >> 10, k = idx & 1023;
    const float* hrow = hidden + (size_t)b * H_DIM;
    const float* wrow = W + (size_t)k * (2 * H_DIM);
    float s = 0.f;
    #pragma unroll 4
    for (int h = 0; h < H_DIM; ++h) s += hrow[h] * wrow[h];
    s1[idx] = s + bias[k];
}

// ---- main: fused GEMM + tanh + v-dot ---------------------------------------
// grid = B_DIM * (T_DIM/TQ) = 512 workgroups of 256 threads (8 wave32).
// LDS: 128 enc rows x 1024 h as bf16, pre-swizzled to WMMA A-fragment order.
// Wave w: m-tiles {2*(w&3), 2*(w&3)+1}, n-tiles [(w>>2)*32, (w>>2)*32+32).
__global__ void attn_energy_kernel(const float* __restrict__ enc,   // (T,B,H)
                                   const float* __restrict__ s1,    // (B,K)
                                   const unsigned short* __restrict__ W2b, // (K,H) bf16
                                   const float* __restrict__ v,     // (K)
                                   float* __restrict__ epart) {     // (2,B,T)
    extern __shared__ char smem[];                 // 256 KB
    const int wg  = blockIdx.x;
    const int b   = wg >> 4;                       // T_DIM/TQ == 16
    const int t0  = (wg & 15) * TQ;
    const int tid = threadIdx.x;
    const int lane = tid & 31;
    const int wave = tid >> 5;                     // 0..7
    const int wmt  = wave & 3;                     // m-tile pair owner
    const int nhalf = wave >> 2;                   // which half of the N range

    // --- stage A block: enc rows (b, t0..t0+127) -> bf16 fragment layout ---
    // chunk = 8 consecutive h of one row -> 16 contiguous LDS bytes.
    for (int c8 = tid; c8 < TQ * (H_DIM / 8); c8 += 256) {
        int m   = c8 >> 7;                 // local row 0..127
        int hc  = c8 & 127;                // 8-elem chunk within row
        int h   = hc * 8;
        int ks  = hc >> 2;                 // k-step (32 h per WMMA)
        int kk8 = hc & 3;                  // which 8-group inside the 32
        const float* src = enc + ((size_t)(t0 + m) * B_DIM + b) * H_DIM + h;
        float4 f0 = *(const float4*)(src);
        float4 f1 = *(const float4*)(src + 4);
        uint4 pk;
        pk.x = pack2(f0.x, f0.y);
        pk.y = pack2(f0.z, f0.w);
        pk.z = pack2(f1.x, f1.y);
        pk.w = pack2(f1.z, f1.w);
        int m_in = m & 15, wt = m >> 4;
        // contiguous A layout: lanes 0-15 hold K {0..15}, lanes 16-31 hold {16..31}
        char* dst = smem + ((size_t)(wt * 32 + ks) * 1024)
                         + (m_in + (kk8 >> 1) * 16) * 32
                         + (kk8 & 1) * 16;
        *(uint4*)dst = pk;
    }
    __syncthreads();

    const int ncol  = lane & 15;           // output column within n-tile
    const int khalf = lane >> 4;           // which K-half this lane carries for B
    const char* aBase0 = smem + (size_t)(wmt * 2)     * 32 * 1024 + lane * 32;
    const char* aBase1 = smem + (size_t)(wmt * 2 + 1) * 32 * 1024 + lane * 32;

    float acc0[8], acc1[8];
    #pragma unroll
    for (int i = 0; i < 8; ++i) { acc0[i] = 0.f; acc1[i] = 0.f; }

    const int nt0 = nhalf * 32;
    for (int nt = nt0; nt < nt0 + 32; ++nt) {      // 32 n-tiles per wave
        const int n = nt * 16 + ncol;
        const float s1v = s1[(size_t)b * K_DIM + n];
        const float vv  = v[n];
        // B fragment: lane ncol = column n; this lane's 16 contiguous h values
        const unsigned short* bp = W2b + (size_t)n * H_DIM + khalf * 16;
        __builtin_prefetch(bp + 16 * H_DIM, 0, 1); // warm next n-tile row

        // split-K: 4 independent WMMA accumulation chains per wave
        v8f c0a = {}, c0b = {}, c1a = {}, c1b = {};
        #pragma unroll 4
        for (int ks2 = 0; ks2 < H_DIM / 64; ++ks2) {   // 16 double k-steps
            v16bf bm0 = *(const v16bf*)(bp + (ks2 * 2)     * 32);
            v16bf bm1 = *(const v16bf*)(bp + (ks2 * 2 + 1) * 32);
            v16bf a00 = *(const v16bf*)(aBase0 + (ks2 * 2)     * 1024);
            v16bf a01 = *(const v16bf*)(aBase0 + (ks2 * 2 + 1) * 1024);
            v16bf a10 = *(const v16bf*)(aBase1 + (ks2 * 2)     * 1024);
            v16bf a11 = *(const v16bf*)(aBase1 + (ks2 * 2 + 1) * 1024);
            c0a = __builtin_amdgcn_wmma_f32_16x16x32_bf16(
                      false, a00, false, bm0, (short)0, c0a, false, false);
            c1a = __builtin_amdgcn_wmma_f32_16x16x32_bf16(
                      false, a10, false, bm0, (short)0, c1a, false, false);
            c0b = __builtin_amdgcn_wmma_f32_16x16x32_bf16(
                      false, a01, false, bm1, (short)0, c0b, false, false);
            c1b = __builtin_amdgcn_wmma_f32_16x16x32_bf16(
                      false, a11, false, bm1, (short)0, c1b, false, false);
        }
        // fuse: tanh + v-weighted reduction over k. C layout: VGPR i holds
        // M = i + 8*khalf, N = ncol.
        #pragma unroll
        for (int i = 0; i < 8; ++i) {
            acc0[i] += fast_tanh(c0a[i] + c0b[i] + s1v) * vv;
            acc1[i] += fast_tanh(c1a[i] + c1b[i] + s1v) * vv;
        }
    }

    // reduce over the 16 lanes (= 16 n columns) of each C half
    #pragma unroll
    for (int i = 0; i < 8; ++i) {
        float x = acc0[i], y = acc1[i];
        x += __shfl_xor(x, 1, 32);  y += __shfl_xor(y, 1, 32);
        x += __shfl_xor(x, 2, 32);  y += __shfl_xor(y, 2, 32);
        x += __shfl_xor(x, 4, 32);  y += __shfl_xor(y, 4, 32);
        x += __shfl_xor(x, 8, 32);  y += __shfl_xor(y, 8, 32);
        acc0[i] = x; acc1[i] = y;
    }
    if (ncol == 0) {
        float* eslab = epart + (size_t)nhalf * B_DIM * T_DIM + (size_t)b * T_DIM;
        const int trow0 = t0 + (wmt * 2)     * 16 + khalf * 8;
        const int trow1 = t0 + (wmt * 2 + 1) * 16 + khalf * 8;
        #pragma unroll
        for (int i = 0; i < 8; ++i) {
            eslab[trow0 + i] = acc0[i];
            eslab[trow1 + i] = acc1[i];
        }
    }
}

// ---- softmax over T per batch row (sums the two N-half partials) -----------
__global__ void softmax_kernel(const float* __restrict__ epart,
                               float* __restrict__ out) {
    __shared__ float red[256];
    const int b = blockIdx.x, tid = threadIdx.x;
    const float* e0 = epart + (size_t)b * T_DIM;
    const float* e1 = epart + (size_t)B_DIM * T_DIM + (size_t)b * T_DIM;

    float mx = -INFINITY;
    for (int t = tid; t < T_DIM; t += 256) mx = fmaxf(mx, e0[t] + e1[t]);
    red[tid] = mx; __syncthreads();
    for (int s = 128; s > 0; s >>= 1) {
        if (tid < s) red[tid] = fmaxf(red[tid], red[tid + s]);
        __syncthreads();
    }
    mx = red[0]; __syncthreads();

    float sum = 0.f;
    for (int t = tid; t < T_DIM; t += 256) sum += __expf(e0[t] + e1[t] - mx);
    red[tid] = sum; __syncthreads();
    for (int s = 128; s > 0; s >>= 1) {
        if (tid < s) red[tid] += red[tid + s];
        __syncthreads();
    }
    const float inv = 1.0f / red[0];
    for (int t = tid; t < T_DIM; t += 256)
        out[(size_t)b * T_DIM + t] = __expf(e0[t] + e1[t] - mx) * inv;
}

// ---------------------------------------------------------------------------
extern "C" void kernel_launch(void* const* d_in, const int* in_sizes, int n_in,
                              void* d_out, int out_size, void* d_ws, size_t ws_size,
                              hipStream_t stream) {
    const float* hidden = (const float*)d_in[0];   // (B,H)
    const float* enc    = (const float*)d_in[1];   // (T,B,H)
    const float* W      = (const float*)d_in[2];   // (H,2H)
    const float* bias   = (const float*)d_in[3];   // (H)
    const float* v      = (const float*)d_in[4];   // (H)
    float* out = (float*)d_out;                    // (B,1,T) flat

    char* ws = (char*)d_ws;
    unsigned short* W2b = (unsigned short*)ws;                               // 2 MB
    float* s1    = (float*)(ws + (size_t)K_DIM * H_DIM * 2);                 // 128 KB
    float* epart = (float*)(ws + (size_t)K_DIM * H_DIM * 2
                               + (size_t)B_DIM * K_DIM * 4);                 // 512 KB

    w2_to_bf16_kernel<<<(K_DIM * H_DIM) / 256, 256, 0, stream>>>(W, W2b);
    s1_kernel<<<(B_DIM * K_DIM) / 256, 256, 0, stream>>>(hidden, W, bias, s1);

    const size_t ldsBytes = (size_t)TQ * H_DIM * 2;        // 256 KB (<= 320 KB/WG)
    attn_energy_kernel<<<B_DIM * (T_DIM / TQ), 256, ldsBytes, stream>>>(
        enc, s1, W2b, v, epart);

    softmax_kernel<<<B_DIM, 256, 0, stream>>>(epart, out);
}
